// Attention_82145544504106
// MI455X (gfx1250) — matrix operand order repaired and checked by reading, not compiled, not run on hardware
//
#include <hip/hip_runtime.h>

#define B_    4
#define N_    2048
#define DIM_  1024
#define H_    16
#define DH_   64
#define TOK   (B_ * N_)        // 8192
#define NQKV  (3 * DIM_)       // 3072
#define SCALE_ 0.125f          // 64^-0.5

typedef __attribute__((ext_vector_type(16))) __bf16        v16bf;
typedef __attribute__((ext_vector_type(8)))  float         v8f;
typedef __attribute__((ext_vector_type(4)))  unsigned int  u32x4;
typedef __attribute__((ext_vector_type(2)))  unsigned int  u32x2;

union Frag {
  v16bf v;
  u32x4 q[2];
  unsigned short s[16];
};

__device__ __forceinline__ unsigned short f2bf(float f) {
  unsigned int u = __float_as_uint(f);
  u += 0x7FFFu + ((u >> 16) & 1u);   // round-to-nearest-even
  return (unsigned short)(u >> 16);
}

__device__ __forceinline__ unsigned int f2bf_pk(float lo, float hi) {
  return (unsigned)f2bf(lo) | ((unsigned)f2bf(hi) << 16);
}

__device__ __forceinline__ v8f wmma_bf16(const Frag& a, const Frag& b, v8f c) {
  return __builtin_amdgcn_wmma_f32_16x16x32_bf16(
      false, a.v, false, b.v, (short)0, c, false, false);
}

// ---------------------------------------------------------------------------
// Kernel 1: LayerNorm (f32 in) -> bf16 xn [TOK, DIM]
// ---------------------------------------------------------------------------
__global__ void __launch_bounds__(256)
ln_kernel(const float* __restrict__ x, const float* __restrict__ g,
          const float* __restrict__ bt, unsigned short* __restrict__ xn) {
  int row  = blockIdx.x;
  int tid  = threadIdx.x;
  int lane = tid & 31, wave = tid >> 5;

  float4 v = ((const float4*)(x + (size_t)row * DIM_))[tid];
  float s  = v.x + v.y + v.z + v.w;
  float ss = v.x * v.x + v.y * v.y + v.z * v.z + v.w * v.w;
  #pragma unroll
  for (int o = 16; o >= 1; o >>= 1) {
    s  += __shfl_xor(s,  o, 32);
    ss += __shfl_xor(ss, o, 32);
  }
  __shared__ float red[2][8];
  if (lane == 0) { red[0][wave] = s; red[1][wave] = ss; }
  __syncthreads();
  if (wave == 0) {
    float a = (lane < 8) ? red[0][lane] : 0.f;
    float c = (lane < 8) ? red[1][lane] : 0.f;
    #pragma unroll
    for (int o = 4; o >= 1; o >>= 1) {
      a += __shfl_xor(a, o, 32);
      c += __shfl_xor(c, o, 32);
    }
    if (lane == 0) { red[0][0] = a; red[1][0] = c; }
  }
  __syncthreads();
  float mean = red[0][0] * (1.0f / DIM_);
  float var  = red[1][0] * (1.0f / DIM_) - mean * mean;
  float rstd = rsqrtf(var + 1e-5f);

  float4 gv = ((const float4*)g )[tid];
  float4 bv = ((const float4*)bt)[tid];
  float y0 = (v.x - mean) * rstd * gv.x + bv.x;
  float y1 = (v.y - mean) * rstd * gv.y + bv.y;
  float y2 = (v.z - mean) * rstd * gv.z + bv.z;
  float y3 = (v.w - mean) * rstd * gv.w + bv.w;

  u32x2 p;
  p.x = f2bf_pk(y0, y1);
  p.y = f2bf_pk(y2, y3);
  ((u32x2*)(xn + (size_t)row * DIM_))[tid] = p;
}

// ---------------------------------------------------------------------------
// Kernel 2: pack f32 weight [K, Ncols] -> bf16 B-fragment layout.
// lane l, elem e -> W[kt*32 + ((l>=16)?16:0) + e][nt*16 + (l&15)]
// ---------------------------------------------------------------------------
__global__ void __launch_bounds__(256)
pack_w(const float* __restrict__ w, unsigned short* __restrict__ out,
       int K, int Ncols) {
  int gid = blockIdx.x * 256 + threadIdx.x;
  int total = (K >> 5) * (Ncols >> 4) * 32;
  if (gid >= total) return;
  int lane = gid & 31;
  int tile = gid >> 5;
  int ntiles = Ncols >> 4;
  int kt = tile / ntiles, nt = tile - kt * ntiles;
  int kb  = kt * 32 + ((lane >> 4) & 1) * 16;
  int col = nt * 16 + (lane & 15);
  Frag f;
  #pragma unroll
  for (int e = 0; e < 16; ++e)
    f.s[e] = f2bf(w[(size_t)(kb + e) * Ncols + col]);
  u32x4* op = (u32x4*)(out + (size_t)gid * 16);
  op[0] = f.q[0];
  op[1] = f.q[1];
}

// ---------------------------------------------------------------------------
// Shared GEMM mainloop: 32(M) x 64(N) wave tile, K-step 32.
// All 12 b128 loads issued before the 8 WMMAs -> partial loadcnt waits,
// compiler software-pipelines across K-steps.
// ---------------------------------------------------------------------------
__device__ __forceinline__ void
gemm_tile_32x64(const unsigned short* __restrict__ arow0,   // row mbase+ml
                const unsigned short* __restrict__ wp,      // packed B
                int ntiles_n, int ntg0, int lane,
                v8f acc[2][4]) {
  int hi = (lane >> 4) & 1;
  const unsigned short* arow1 = arow0 + 16 * DIM_;
  #pragma unroll 1
  for (int kt = 0; kt < DIM_ / 32; ++kt) {
    int ka = kt * 32 + hi * 16;
    Frag a0, a1, b0, b1, b2, b3;
    a0.q[0] = *(const u32x4*)(arow0 + ka);
    a0.q[1] = *(const u32x4*)(arow0 + ka + 8);
    a1.q[0] = *(const u32x4*)(arow1 + ka);
    a1.q[1] = *(const u32x4*)(arow1 + ka + 8);
    const unsigned short* bp =
        wp + ((size_t)(kt * ntiles_n + ntg0) * 32 + lane) * 16;
    b0.q[0] = ((const u32x4*)bp)[0];  b0.q[1] = ((const u32x4*)bp)[1];
    bp += 32 * 16;
    b1.q[0] = ((const u32x4*)bp)[0];  b1.q[1] = ((const u32x4*)bp)[1];
    bp += 32 * 16;
    b2.q[0] = ((const u32x4*)bp)[0];  b2.q[1] = ((const u32x4*)bp)[1];
    bp += 32 * 16;
    b3.q[0] = ((const u32x4*)bp)[0];  b3.q[1] = ((const u32x4*)bp)[1];

    // speculative prefetch of next A stream (OOB prefetch is dropped by HW)
    __builtin_prefetch(arow0 + ka + 32, 0, 3);
    __builtin_prefetch(arow1 + ka + 32, 0, 3);

    acc[0][0] = wmma_bf16(a0, b0, acc[0][0]);
    acc[1][0] = wmma_bf16(a1, b0, acc[1][0]);
    acc[0][1] = wmma_bf16(a0, b1, acc[0][1]);
    acc[1][1] = wmma_bf16(a1, b1, acc[1][1]);
    acc[0][2] = wmma_bf16(a0, b2, acc[0][2]);
    acc[1][2] = wmma_bf16(a1, b2, acc[1][2]);
    acc[0][3] = wmma_bf16(a0, b3, acc[0][3]);
    acc[1][3] = wmma_bf16(a1, b3, acc[1][3]);
  }
}

// ---------------------------------------------------------------------------
// Kernel 3: QKV GEMM: xn[TOK,DIM] @ wqkv_packed -> Q,K [b,h,n,d] bf16,
//           V transposed [b,h,d,n] bf16. Wave tile 32x64, block 128x128.
// ---------------------------------------------------------------------------
__global__ void __launch_bounds__(256)
qkv_gemm(const unsigned short* __restrict__ xn,
         const unsigned short* __restrict__ wp,
         const float* __restrict__ bias,
         unsigned short* __restrict__ qb,
         unsigned short* __restrict__ kb,
         unsigned short* __restrict__ vT) {
  int tid  = threadIdx.x;
  int lane = tid & 31, wave = tid >> 5;
  int hi = (lane >> 4) & 1, ml = lane & 15;
  int mw = wave & 3, nw = wave >> 2;
  int mbase = blockIdx.x * 128 + mw * 32;
  int nbase = blockIdx.y * 128 + nw * 64;
  int ntg0  = nbase >> 4;

  v8f acc[2][4] = {};
  gemm_tile_32x64(xn + (size_t)(mbase + ml) * DIM_, wp, NQKV / 16, ntg0,
                  lane, acc);

  int rbase = hi * 8;
  #pragma unroll
  for (int t = 0; t < 4; ++t) {
    int colbase = nbase + t * 16;       // uniform across lanes
    int which   = colbase >> 10;        // uniform: scalar branch, no EXEC mask
    int inner0  = colbase & 1023;
    int head    = inner0 >> 6;          // uniform (16-tile never crosses 64)
    int dh      = (inner0 & 63) + ml;
    float bv    = bias[colbase + ml];
    #pragma unroll
    for (int i = 0; i < 2; ++i) {
      int m0 = mbase + i * 16 + rbase;  // 8 consecutive rows, same batch
      int bb = m0 >> 11;
      int nn0 = m0 & 2047;
      size_t bh = (size_t)(bb * H_ + head);
      if (which == 2) {
        // V transposed: consecutive rows -> consecutive bf16 -> pack b32
        unsigned int* vp =
            (unsigned int*)(vT + (bh * DH_ + dh) * N_ + nn0);
        #pragma unroll
        for (int r = 0; r < 4; ++r)
          vp[r] = f2bf_pk(acc[i][t][2 * r], acc[i][t][2 * r + 1]);
      } else {
        unsigned short* dst = (which == 0) ? qb : kb;
        dst += (bh * N_ + nn0) * DH_ + dh;
        #pragma unroll
        for (int r = 0; r < 8; ++r)
          dst[(size_t)r * DH_] = f2bf(acc[i][t][r] + bv);
      }
    }
  }
}

// ---------------------------------------------------------------------------
// Kernel 4: flash attention. One 16-query tile per wave, 32 keys per step.
// All K and V fragment loads issued up front so S-WMMAs + softmax + LDS
// round-trip overlap the V loads.
// ---------------------------------------------------------------------------
__global__ void __launch_bounds__(256)
attn_kernel(const unsigned short* __restrict__ qb,
            const unsigned short* __restrict__ kbuf,
            const unsigned short* __restrict__ vT,
            unsigned short* __restrict__ ob) {
  __shared__ unsigned short sm[8 * 512];   // per-wave 16x32 bf16 P tile
  int tid  = threadIdx.x;
  int lane = tid & 31, wave = tid >> 5;
  int hi = (lane >> 4) & 1, ml = lane & 15;

  int tileid = blockIdx.x * 8 + wave;
  int bh = tileid >> 7;          // / (2048/16)
  int qt = tileid & 127;

  const unsigned short* Qp = qb   + ((size_t)bh * N_ + qt * 16) * DH_;
  const unsigned short* Kp = kbuf + (size_t)bh * N_ * DH_;
  const unsigned short* Vp = vT   + (size_t)bh * DH_ * N_;
  unsigned short* lds = sm + wave * 512;

  Frag q0, q1;                    // Q A-fragments, depth chunks 0..31 / 32..63
  {
    const unsigned short* qr = Qp + ml * DH_ + hi * 16;
    q0.q[0] = *(const u32x4*)(qr);      q0.q[1] = *(const u32x4*)(qr + 8);
    q1.q[0] = *(const u32x4*)(qr + 32); q1.q[1] = *(const u32x4*)(qr + 40);
  }

  float mi[8], li[8];
  #pragma unroll
  for (int r = 0; r < 8; ++r) { mi[r] = -3.0e38f; li[r] = 0.f; }
  v8f o0 = {}, o1 = {}, o2 = {}, o3 = {};

  #pragma unroll 1
  for (int kt = 0; kt < N_ / 32; ++kt) {
    int j0 = kt * 32;

    // ---- issue ALL loads for this step (8 K-chunks + 8 V-chunks) ----
    Frag ka0, ka1, kc0, kc1;     // S0 chunks (keys j0..15), S1 (j0+16..31)
    {
      const unsigned short* kr = Kp + (size_t)(j0 + ml) * DH_ + hi * 16;
      ka0.q[0] = *(const u32x4*)(kr);      ka0.q[1] = *(const u32x4*)(kr + 8);
      ka1.q[0] = *(const u32x4*)(kr + 32); ka1.q[1] = *(const u32x4*)(kr + 40);
    }
    {
      const unsigned short* kr = Kp + (size_t)(j0 + 16 + ml) * DH_ + hi * 16;
      kc0.q[0] = *(const u32x4*)(kr);      kc0.q[1] = *(const u32x4*)(kr + 8);
      kc1.q[0] = *(const u32x4*)(kr + 32); kc1.q[1] = *(const u32x4*)(kr + 40);
    }
    Frag vf0, vf1, vf2, vf3;
    {
      const unsigned short* vr = Vp + (size_t)ml * N_ + j0 + hi * 16;
      vf0.q[0] = *(const u32x4*)(vr);  vf0.q[1] = *(const u32x4*)(vr + 8);
      vr += 16 * N_;
      vf1.q[0] = *(const u32x4*)(vr);  vf1.q[1] = *(const u32x4*)(vr + 8);
      vr += 16 * N_;
      vf2.q[0] = *(const u32x4*)(vr);  vf2.q[1] = *(const u32x4*)(vr + 8);
      vr += 16 * N_;
      vf3.q[0] = *(const u32x4*)(vr);  vf3.q[1] = *(const u32x4*)(vr + 8);
    }

    // ---- S = Q @ K^T ----
    v8f s0 = {}, s1 = {};
    s0 = wmma_bf16(q0, ka0, s0);
    s0 = wmma_bf16(q1, ka1, s0);
    s1 = wmma_bf16(q0, kc0, s1);
    s1 = wmma_bf16(q1, kc1, s1);

    // ---- online softmax; row r lives on 16 lanes of this half-wave ----
    #pragma unroll
    for (int r = 0; r < 8; ++r) {
      float a0 = s0[r] * SCALE_, a1 = s1[r] * SCALE_;
      float mx = fmaxf(a0, a1);
      mx = fmaxf(mx, __shfl_xor(mx, 1, 32));
      mx = fmaxf(mx, __shfl_xor(mx, 2, 32));
      mx = fmaxf(mx, __shfl_xor(mx, 4, 32));
      mx = fmaxf(mx, __shfl_xor(mx, 8, 32));
      float nm = fmaxf(mi[r], mx);
      float al = __expf(mi[r] - nm);
      float p0 = __expf(a0 - nm);
      float p1 = __expf(a1 - nm);
      float rs = p0 + p1;
      rs += __shfl_xor(rs, 1, 32);
      rs += __shfl_xor(rs, 2, 32);
      rs += __shfl_xor(rs, 4, 32);
      rs += __shfl_xor(rs, 8, 32);
      li[r] = li[r] * al + rs;
      mi[r] = nm;
      o0[r] *= al; o1[r] *= al; o2[r] *= al; o3[r] *= al;
      int row = hi * 8 + r;                 // C-layout -> LDS 16x32 row-major
      lds[row * 32 + ml]      = f2bf(p0);
      lds[row * 32 + 16 + ml] = f2bf(p1);
    }

    // reload P as A-fragment (wave-private LDS; DS ops are wave-ordered)
    Frag pf;
    {
      const unsigned short* pr = lds + ml * 32 + hi * 16;
      pf.q[0] = *(const u32x4*)(pr);
      pf.q[1] = *(const u32x4*)(pr + 8);
    }

    // ---- O += P @ V (V already in registers) ----
    o0 = wmma_bf16(pf, vf0, o0);
    o1 = wmma_bf16(pf, vf1, o1);
    o2 = wmma_bf16(pf, vf2, o2);
    o3 = wmma_bf16(pf, vf3, o3);
  }

  int b = bh >> 4, h = bh & 15;
  #pragma unroll
  for (int r = 0; r < 8; ++r) {
    float inv = 1.0f / li[r];
    int token = b * N_ + qt * 16 + hi * 8 + r;
    unsigned short* orow = ob + (size_t)token * (H_ * DH_) + h * DH_ + ml;
    orow[0]  = f2bf(o0[r] * inv);
    orow[16] = f2bf(o1[r] * inv);
    orow[32] = f2bf(o2[r] * inv);
    orow[48] = f2bf(o3[r] * inv);
  }
}

// ---------------------------------------------------------------------------
// Kernel 5: projection GEMM: o[TOK,1024] @ wproj -> f32 out + bias
// ---------------------------------------------------------------------------
__global__ void __launch_bounds__(256)
proj_gemm(const unsigned short* __restrict__ ab,
          const unsigned short* __restrict__ wp,
          const float* __restrict__ bias,
          float* __restrict__ out) {
  int tid  = threadIdx.x;
  int lane = tid & 31, wave = tid >> 5;
  int hi = (lane >> 4) & 1, ml = lane & 15;
  int mw = wave & 3, nw = wave >> 2;
  int mbase = blockIdx.x * 128 + mw * 32;
  int nbase = blockIdx.y * 128 + nw * 64;
  int ntg0  = nbase >> 4;

  v8f acc[2][4] = {};
  gemm_tile_32x64(ab + (size_t)(mbase + ml) * DIM_, wp, DIM_ / 16, ntg0,
                  lane, acc);

  int rbase = hi * 8;
  #pragma unroll
  for (int t = 0; t < 4; ++t) {
    int ncol = nbase + t * 16 + ml;
    float bv = bias[ncol];
    #pragma unroll
    for (int i = 0; i < 2; ++i) {
      #pragma unroll
      for (int r = 0; r < 8; ++r) {
        int m = mbase + i * 16 + rbase + r;
        out[(size_t)m * DIM_ + ncol] = acc[i][t][r] + bv;
      }
    }
  }
}

// ---------------------------------------------------------------------------
extern "C" void kernel_launch(void* const* d_in, const int* in_sizes, int n_in,
                              void* d_out, int out_size, void* d_ws, size_t ws_size,
                              hipStream_t stream) {
  (void)in_sizes; (void)n_in; (void)out_size; (void)ws_size;
  const float* x     = (const float*)d_in[0];
  const float* gam   = (const float*)d_in[1];
  const float* bet   = (const float*)d_in[2];
  const float* wqkv  = (const float*)d_in[3];
  const float* bqkv  = (const float*)d_in[4];
  const float* wproj = (const float*)d_in[5];
  const float* bproj = (const float*)d_in[6];
  float* out = (float*)d_out;

  char* ws = (char*)d_ws;
  size_t off = 0;
  unsigned short* xn  = (unsigned short*)(ws + off); off += (size_t)TOK  * DIM_ * 2;
  unsigned short* wqp = (unsigned short*)(ws + off); off += (size_t)DIM_ * NQKV * 2;
  unsigned short* wpp = (unsigned short*)(ws + off); off += (size_t)DIM_ * DIM_ * 2;
  unsigned short* qb  = (unsigned short*)(ws + off); off += (size_t)TOK  * DIM_ * 2;
  unsigned short* kb  = (unsigned short*)(ws + off); off += (size_t)TOK  * DIM_ * 2;
  unsigned short* vT  = (unsigned short*)(ws + off); off += (size_t)TOK  * DIM_ * 2;
  unsigned short* ob  = (unsigned short*)(ws + off); off += (size_t)TOK  * DIM_ * 2;

  ln_kernel<<<TOK, 256, 0, stream>>>(x, gam, bet, xn);
  pack_w<<<(DIM_ * NQKV / 16 + 255) / 256, 256, 0, stream>>>(wqkv, wqp, DIM_, NQKV);
  pack_w<<<(DIM_ * DIM_ / 16 + 255) / 256, 256, 0, stream>>>(wproj, wpp, DIM_, DIM_);
  qkv_gemm<<<dim3(TOK / 128, NQKV / 128), 256, 0, stream>>>(xn, wqp, bqkv, qb, kb, vT);
  attn_kernel<<<(B_ * H_ * (N_ / 16)) / 8, 256, 0, stream>>>(qb, kb, vT, ob);
  proj_gemm<<<dim3(TOK / 128, DIM_ / 128), 256, 0, stream>>>(ob, wpp, bproj, out);
}
